// CrossAttentionFuse_12627203850515
// MI455X (gfx1250) — matrix operand order repaired and checked by reading, not compiled, not run on hardware
//
#include <hip/hip_runtime.h>
#include <hip/hip_bf16.h>
#include <math.h>

// ---------------- problem constants ----------------
#define B_ROWS 32768
#define DQ 1536
#define DK 768
#define DM 256
#define DFF 512
#define NT 16          // 16 N-tiles of 16 cols = 256 cols per wave
#define WPB 8          // waves per block (256 threads, wave32)
#define BLOCK_ROWS 128 // 8 waves * 16 rows
#define BPITCH 40      // lds weight-tile row pitch (bf16 elems): 64B data + 16B pad
#define APITCH 264     // per-wave A-scratch pitch (256 + pad)
#define HPITCH 520     // per-wave FFN-hidden pitch (512 + pad)
#define LDSB_USH (256 * BPITCH)
#define LDSA_USH (16 * APITCH)
#define LDSH_USH (16 * HPITCH)
// 4 weight-staging buffers (double-buffered, 2 matrices in phase A)
#define SMEM_USH (4 * LDSB_USH + WPB * LDSA_USH + WPB * LDSH_USH)
#define SMEM_BYTES (SMEM_USH * 2)   // 282,624 B <= 320KB WGP LDS

typedef __attribute__((ext_vector_type(16))) __bf16 v16bf;
typedef __attribute__((ext_vector_type(8)))  float  v8f;
typedef unsigned int v4u  __attribute__((ext_vector_type(4)));
typedef int          v4i  __attribute__((ext_vector_type(4)));
typedef int          v8i  __attribute__((ext_vector_type(8)));

#if __has_builtin(__builtin_amdgcn_tensor_load_to_lds) && \
    __has_builtin(__builtin_amdgcn_s_wait_tensorcnt)
#define HAVE_TDM 1
#else
#define HAVE_TDM 0
#endif

struct alignas(16) U128 { unsigned int x[4]; };

struct V16 {
  union {
    v16bf v;
    unsigned short us[16];
    U128 q[2];
  };
};

__device__ __forceinline__ unsigned short f2bf(float x) {
  unsigned u = __float_as_uint(x);
  return (unsigned short)((u + 0x7FFFu + ((u >> 16) & 1u)) >> 16);
}

__device__ __forceinline__ v8f wmma_bf16(v16bf a, v16bf b, v8f c) {
  // D = A(16x32 bf16) * B(32x16 bf16) + C(16x16 f32)
  return __builtin_amdgcn_wmma_f32_16x16x32_bf16(false, a, false, b,
                                                 (short)0, c, false, false);
}

__device__ __forceinline__ void zero_acc(v8f (&a)[NT]) {
#pragma unroll
  for (int n = 0; n < NT; ++n)
#pragma unroll
    for (int j = 0; j < 8; ++j) a[n][j] = 0.f;
}

// 16-lane (half-wave) sum reduction; wave32, masks stay within 16-lane groups.
__device__ __forceinline__ float hred16(float x) {
  x += __shfl_xor(x, 1);
  x += __shfl_xor(x, 2);
  x += __shfl_xor(x, 4);
  x += __shfl_xor(x, 8);
  return x;
}

#if HAVE_TDM
// Issue one TDM descriptor: load weight tile rows 0..255, cols kc..kc+31 (bf16)
// of a row-major [256 x ldK] matrix into LDS with BPITCH-padded rows.
// D# per CDNA5 ISA ch.8: 2-D tile, data_size=2B, pad 4 DWORDs every 16 DWORDs.
__device__ __forceinline__ void tdm_stage(const unsigned short* W, int ldK, int kc,
                                          unsigned short* ldsDst) {
  unsigned long long ga = (unsigned long long)(const void*)(W + kc); // byte addr
  unsigned lds_off = (unsigned)(unsigned long long)(void*)ldsDst;    // LDS byte addr
  v4u g0;
  g0[0] = 1u;                                   // count=1, no gather
  g0[1] = lds_off;                              // lds_addr
  g0[2] = (unsigned)ga;                         // global_addr[31:0]
  g0[3] = (unsigned)((ga >> 32) & 0x01FFFFFFu)  // global_addr[56:32]
          | (2u << 30);                         // type=2 ("image")
  unsigned s1 = (unsigned)(ldK * 256);          // tensor_dim1_stride (elems)
  v8i g1;
  g1[0] = (int)((1u << 16)        // data_size = 2B
              | (1u << 20)        // pad_enable
              | (3u << 22)        // pad_interval: 16 DWORDs (one 64B row)
              | (3u << 25));      // pad_amount: 4 DWORDs (16B) -> 80B pitch
  g1[1] = (int)(((unsigned)ldK & 0xFFFFu) << 16);       // tensor_dim0[15:0]
  g1[2] = (int)((256u << 16) | ((unsigned)ldK >> 16));  // td0[31:16] | td1[15:0]
  g1[3] = (int)(32u << 16);                             // tile_dim0 = 32
  g1[4] = 256;                                          // tile_dim1 = 256
  g1[5] = ldK;                                          // tensor_dim0_stride[31:0]
  g1[6] = (int)((s1 & 0xFFFFu) << 16);                  // d0s[47:32]=0 | d1s[15:0]
  g1[7] = (int)(s1 >> 16);                              // d1s[47:16]
  v4i gz; gz[0] = 0; gz[1] = 0; gz[2] = 0; gz[3] = 0;   // 2-D tile: groups 2/3 unused
#if __clang_major__ >= 23
  v8i gz8; gz8[0]=0; gz8[1]=0; gz8[2]=0; gz8[3]=0; gz8[4]=0; gz8[5]=0; gz8[6]=0; gz8[7]=0;
  __builtin_amdgcn_tensor_load_to_lds(g0, g1, gz, gz, gz8, 0);
#else
  __builtin_amdgcn_tensor_load_to_lds(g0, g1, gz, gz, 0);
#endif
}
#endif // HAVE_TDM

// Fallback cooperative stage: thread t copies weight row t (32 bf16 = 4 x b128).
__device__ __forceinline__ void stage_w(const unsigned short* __restrict__ W,
                                        int ldK, int kc,
                                        unsigned short* lds, int tid) {
  const U128* g = (const U128*)(W + (size_t)tid * ldK + kc);
  U128* s = (U128*)(lds + tid * BPITCH);
  s[0] = g[0]; s[1] = g[1]; s[2] = g[2]; s[3] = g[3];
}

// A operand (16x32 bf16) from row-major f32 global, converted on the fly.
// lane<16: row=lane, K=[0..7],[16..23]; lane>=16: row=lane-16, K=[8..15],[24..31]
__device__ __forceinline__ v16bf load_a_global(const float* __restrict__ src,
                                               int ld, int row0, int kc, int lane) {
  const int off0 = (lane < 16) ? 0 : 8;
  const float* p = src + (size_t)(row0 + (lane & 15)) * ld + kc + off0;
  float4 f0 = *(const float4*)(p);
  float4 f1 = *(const float4*)(p + 4);
  float4 f2 = *(const float4*)(p + 16);
  float4 f3 = *(const float4*)(p + 20);
  V16 a;
  a.us[0]  = f2bf(f0.x); a.us[1]  = f2bf(f0.y); a.us[2]  = f2bf(f0.z); a.us[3]  = f2bf(f0.w);
  a.us[4]  = f2bf(f1.x); a.us[5]  = f2bf(f1.y); a.us[6]  = f2bf(f1.z); a.us[7]  = f2bf(f1.w);
  a.us[8]  = f2bf(f2.x); a.us[9]  = f2bf(f2.y); a.us[10] = f2bf(f2.z); a.us[11] = f2bf(f2.w);
  a.us[12] = f2bf(f3.x); a.us[13] = f2bf(f3.y); a.us[14] = f2bf(f3.z); a.us[15] = f2bf(f3.w);
  return a.v;
}

// A operand from per-wave LDS bf16 scratch (row-major, pitch elems).
__device__ __forceinline__ v16bf load_a_lds(const unsigned short* ldsSrc,
                                            int pitch, int kc, int lane) {
  const int r = lane & 15;
  const int k0 = kc + ((lane < 16) ? 0 : 8);
  V16 a;
  a.q[0] = *(const U128*)(ldsSrc + r * pitch + k0);
  a.q[1] = *(const U128*)(ldsSrc + r * pitch + k0 + 16);
  return a.v;
}

// B operand (32x16 bf16) from staged weight tile: lane<16 -> col N=lane, K=[0..7],[16..23];
// lane>=16 -> col N=lane-16, K=[8..15],[24..31]. Weight stored row-major [N][K].
__device__ __forceinline__ v16bf load_b_lds(const unsigned short* ldsB,
                                            int ntile, int lane) {
  const int r = ntile * 16 + (lane & 15);
  const int k0 = (lane < 16) ? 0 : 8;
  V16 b;
  b.q[0] = *(const U128*)(ldsB + r * BPITCH + k0);
  b.q[1] = *(const U128*)(ldsB + r * BPITCH + k0 + 16);
  return b.v;
}

// GEMM with A sourced from per-wave LDS scratch; weight tiles staged per K-step.
// TDM path: wave0 drives the Tensor Data Mover, double-buffered, 1 barrier/step.
template <int KTOT>
__device__ __forceinline__ void gemm_ldsA(const unsigned short* ldsSrc, int pitch,
                                          const unsigned short* __restrict__ W, int ldK,
                                          unsigned short* ldsB0, unsigned short* ldsB1,
                                          v8f (&acc)[NT], int tid, int wave, int lane) {
#if HAVE_TDM
  if (wave == 0) tdm_stage(W, ldK, 0, ldsB0);
  for (int kc = 0; kc < KTOT; kc += 32) {
    unsigned short* cur = ((kc >> 5) & 1) ? ldsB1 : ldsB0;
    unsigned short* nxt = ((kc >> 5) & 1) ? ldsB0 : ldsB1;
    if (wave == 0) __builtin_amdgcn_s_wait_tensorcnt(0);
    __syncthreads();                        // tile kc visible; tile kc-32 consumed
    if (wave == 0 && kc + 32 < KTOT) tdm_stage(W, ldK, kc + 32, nxt);
    v16bf a = load_a_lds(ldsSrc, pitch, kc, lane);
#pragma unroll
    for (int n = 0; n < NT; ++n)
      acc[n] = wmma_bf16(a, load_b_lds(cur, n, lane), acc[n]);
  }
  __syncthreads();                          // buffers free for next stager
#else
  for (int kc = 0; kc < KTOT; kc += 32) {
    stage_w(W, ldK, kc, ldsB0, tid);
    __syncthreads();
    v16bf a = load_a_lds(ldsSrc, pitch, kc, lane);
#pragma unroll
    for (int n = 0; n < NT; ++n)
      acc[n] = wmma_bf16(a, load_b_lds(ldsB0, n, lane), acc[n]);
    __syncthreads();
  }
  (void)ldsB1;
#endif
}

// Per-row LayerNorm over the 16 tiles held by one wave (rows in VGPR slots).
__device__ __forceinline__ void layernorm_tiles(v8f (&x)[NT],
                                                const float* __restrict__ g,
                                                const float* __restrict__ b,
                                                int ln15) {
  float mean[8], rstd[8];
#pragma unroll
  for (int j = 0; j < 8; ++j) {
    float s = 0.f;
#pragma unroll
    for (int n = 0; n < NT; ++n) s += x[n][j];
    mean[j] = hred16(s) * (1.f / 256.f);
  }
#pragma unroll
  for (int j = 0; j < 8; ++j) {
    float s = 0.f;
#pragma unroll
    for (int n = 0; n < NT; ++n) {
      float d = x[n][j] - mean[j];
      s += d * d;
    }
    rstd[j] = rsqrtf(hred16(s) * (1.f / 256.f) + 1e-5f);
  }
#pragma unroll
  for (int n = 0; n < NT; ++n) {
    const int c = n * 16 + ln15;
    const float gv = g[c], bv = b[c];
#pragma unroll
    for (int j = 0; j < 8; ++j)
      x[n][j] = (x[n][j] - mean[j]) * rstd[j] * gv + bv;
  }
}

// ---------------- weight f32 -> bf16 prep ----------------
__global__ void cvt_bf16_kernel(const float* __restrict__ src,
                                unsigned short* __restrict__ dst, int n) {
  int i = blockIdx.x * blockDim.x + threadIdx.x;
  if (i < n) dst[i] = f2bf(src[i]);
}

// ---------------- fully fused main kernel ----------------
__global__ __launch_bounds__(256)
void fused_kernel(const float* __restrict__ qvec, const float* __restrict__ kvvec,
                  const unsigned short* __restrict__ Wq, const unsigned short* __restrict__ Wg,
                  const unsigned short* __restrict__ Wv, const unsigned short* __restrict__ Wi,
                  const unsigned short* __restrict__ Wo, const unsigned short* __restrict__ W1,
                  const unsigned short* __restrict__ W2,
                  const float* __restrict__ inb, const float* __restrict__ outb,
                  const float* __restrict__ bg,
                  const float* __restrict__ l1g, const float* __restrict__ l1b,
                  const float* __restrict__ l2g, const float* __restrict__ l2b,
                  float* __restrict__ out) {
  extern __shared__ unsigned short smem[];
  const int tid = threadIdx.x;
  const int lane = tid & 31;
  const int wave = tid >> 5;
  const int ln15 = lane & 15;
  const int row0 = blockIdx.x * BLOCK_ROWS + wave * 16;

  unsigned short* ldsBq0 = smem;                 // ping/pong for matrix 0
  unsigned short* ldsBq1 = smem + 1 * LDSB_USH;
  unsigned short* ldsBg0 = smem + 2 * LDSB_USH;  // ping/pong for matrix 1 (phase A)
  unsigned short* ldsBg1 = smem + 3 * LDSB_USH;
  unsigned short* ldsA   = smem + 4 * LDSB_USH + wave * LDSA_USH;
  unsigned short* ldsH   = smem + 4 * LDSB_USH + WPB * LDSA_USH + wave * LDSH_USH;

  v8f qa[NT], ga[NT], xa[NT];  // q / gate / generic accumulators

  // ---- Phase A: q = qvec@Wq.T and gate_pre = qvec@Wg.T (shared A, K=1536)
  zero_acc(qa); zero_acc(ga);
#if HAVE_TDM
  if (wave == 0) { tdm_stage(Wq, DQ, 0, ldsBq0); tdm_stage(Wg, DQ, 0, ldsBg0); }
  for (int kc = 0; kc < DQ; kc += 32) {
    const int pp = (kc >> 5) & 1;
    unsigned short* curQ = pp ? ldsBq1 : ldsBq0;
    unsigned short* curG = pp ? ldsBg1 : ldsBg0;
    if (wave == 0) __builtin_amdgcn_s_wait_tensorcnt(0);
    __syncthreads();
    if (wave == 0 && kc + 32 < DQ) {
      tdm_stage(Wq, DQ, kc + 32, pp ? ldsBq0 : ldsBq1);
      tdm_stage(Wg, DQ, kc + 32, pp ? ldsBg0 : ldsBg1);
    }
    if (kc + 32 < DQ)
      __builtin_prefetch(qvec + (size_t)(row0 + ln15) * DQ + kc + 32, 0, 3);
    v16bf a = load_a_global(qvec, DQ, row0, kc, lane);
#pragma unroll
    for (int n = 0; n < NT; ++n) {
      qa[n] = wmma_bf16(a, load_b_lds(curQ, n, lane), qa[n]);
      ga[n] = wmma_bf16(a, load_b_lds(curG, n, lane), ga[n]);
    }
  }
  __syncthreads();
#else
  for (int kc = 0; kc < DQ; kc += 32) {
    stage_w(Wq, DQ, kc, ldsBq0, tid);
    stage_w(Wg, DQ, kc, ldsBg0, tid);
    if (kc + 32 < DQ)
      __builtin_prefetch(qvec + (size_t)(row0 + ln15) * DQ + kc + 32, 0, 3);
    __syncthreads();
    v16bf a = load_a_global(qvec, DQ, row0, kc, lane);
#pragma unroll
    for (int n = 0; n < NT; ++n) {
      qa[n] = wmma_bf16(a, load_b_lds(ldsBq0, n, lane), qa[n]);
      ga[n] = wmma_bf16(a, load_b_lds(ldsBg0, n, lane), ga[n]);
    }
    __syncthreads();
  }
#endif
  // gate = sigmoid(gate_pre + bg)  (in place)
#pragma unroll
  for (int n = 0; n < NT; ++n) {
    const float bgv = bg[n * 16 + ln15];
#pragma unroll
    for (int j = 0; j < 8; ++j)
      ga[n][j] = 1.f / (1.f + __expf(-(ga[n][j] + bgv)));
  }

  // ---- Phase B: v = kvvec@Wv.T (K=768)
  zero_acc(xa);
#if HAVE_TDM
  if (wave == 0) tdm_stage(Wv, DK, 0, ldsBq0);
  for (int kc = 0; kc < DK; kc += 32) {
    const int pp = (kc >> 5) & 1;
    unsigned short* cur = pp ? ldsBq1 : ldsBq0;
    if (wave == 0) __builtin_amdgcn_s_wait_tensorcnt(0);
    __syncthreads();
    if (wave == 0 && kc + 32 < DK) tdm_stage(Wv, DK, kc + 32, pp ? ldsBq0 : ldsBq1);
    if (kc + 32 < DK)
      __builtin_prefetch(kvvec + (size_t)(row0 + ln15) * DK + kc + 32, 0, 3);
    v16bf a = load_a_global(kvvec, DK, row0, kc, lane);
#pragma unroll
    for (int n = 0; n < NT; ++n)
      xa[n] = wmma_bf16(a, load_b_lds(cur, n, lane), xa[n]);
  }
  __syncthreads();
#else
  for (int kc = 0; kc < DK; kc += 32) {
    stage_w(Wv, DK, kc, ldsBq0, tid);
    if (kc + 32 < DK)
      __builtin_prefetch(kvvec + (size_t)(row0 + ln15) * DK + kc + 32, 0, 3);
    __syncthreads();
    v16bf a = load_a_global(kvvec, DK, row0, kc, lane);
#pragma unroll
    for (int n = 0; n < NT; ++n)
      xa[n] = wmma_bf16(a, load_b_lds(ldsBq0, n, lane), xa[n]);
    __syncthreads();
  }
#endif
  // scatter v (C-layout -> row-major bf16 A-scratch)
#pragma unroll
  for (int n = 0; n < NT; ++n) {
    const int c = n * 16 + ln15;
#pragma unroll
    for (int j = 0; j < 8; ++j) {
      const int r = (lane < 16) ? j : (8 + j);
      ldsA[r * APITCH + c] = f2bf(xa[n][j]);
    }
  }

  // ---- Phase C: vi = v @ in_w[2D:].T + in_b[2D:]   (softmax over len-1 == 1)
  zero_acc(xa);
  gemm_ldsA<DM>(ldsA, APITCH, Wi, DM, ldsBq0, ldsBq1, xa, tid, wave, lane);
#pragma unroll
  for (int n = 0; n < NT; ++n) {
    const int c = n * 16 + ln15;
    const float bi = inb[2 * DM + c];
#pragma unroll
    for (int j = 0; j < 8; ++j) {
      const int r = (lane < 16) ? j : (8 + j);
      ldsA[r * APITCH + c] = f2bf(xa[n][j] + bi);
    }
  }

  // ---- Phase D: attn = vi @ out_w.T (+out_b later)
  zero_acc(xa);
  gemm_ldsA<DM>(ldsA, APITCH, Wo, DM, ldsBq0, ldsBq1, xa, tid, wave, lane);

  // ---- Phase E: y = LN1((attn+out_b)*gate + q); keep y in qa (f32 regs)
#pragma unroll
  for (int n = 0; n < NT; ++n) {
    const float ob = outb[n * 16 + ln15];
#pragma unroll
    for (int j = 0; j < 8; ++j)
      qa[n][j] = (xa[n][j] + ob) * ga[n][j] + qa[n][j];
  }
  layernorm_tiles(qa, l1g, l1b, ln15);
  // scatter y bf16 for FFN A operand
#pragma unroll
  for (int n = 0; n < NT; ++n) {
    const int c = n * 16 + ln15;
#pragma unroll
    for (int j = 0; j < 8; ++j) {
      const int r = (lane < 16) ? j : (8 + j);
      ldsA[r * APITCH + c] = f2bf(qa[n][j]);
    }
  }

  // ---- Phase F1: h = gelu(y @ W1.T), 512 cols in two 256-col halves
  for (int half = 0; half < 2; ++half) {
    zero_acc(xa);
    gemm_ldsA<DM>(ldsA, APITCH, W1 + (size_t)half * 256 * DM, DM,
                  ldsBq0, ldsBq1, xa, tid, wave, lane);
#pragma unroll
    for (int n = 0; n < NT; ++n) {
      const int c = half * 256 + n * 16 + ln15;
#pragma unroll
      for (int j = 0; j < 8; ++j) {
        const int r = (lane < 16) ? j : (8 + j);
        const float v = xa[n][j];
        const float ge = 0.5f * v * (1.f + erff(v * 0.70710678118654752f));
        ldsH[r * HPITCH + c] = f2bf(ge);
      }
    }
  }

  // ---- Phase F2: y2 = h @ W2.T (K=512)
  zero_acc(xa);
  gemm_ldsA<DFF>(ldsH, HPITCH, W2, DFF, ldsBq0, ldsBq1, xa, tid, wave, lane);

  // ---- Phase G: out = LN2(y + y2)
#pragma unroll
  for (int n = 0; n < NT; ++n)
#pragma unroll
    for (int j = 0; j < 8; ++j)
      qa[n][j] += xa[n][j];
  layernorm_tiles(qa, l2g, l2b, ln15);

  __syncthreads();
  // stage result f32 in per-wave LDS, then coalesced b128 stores
  float* ldsf = (float*)ldsH;
#pragma unroll
  for (int n = 0; n < NT; ++n) {
    const int c = n * 16 + ln15;
#pragma unroll
    for (int j = 0; j < 8; ++j) {
      const int r = (lane < 16) ? j : (8 + j);
      ldsf[r * 256 + c] = qa[n][j];
    }
  }
  __syncthreads();
  for (int it = 0; it < 32; ++it) {
    const int fi = it * 32 + lane;      // 1024 float4 chunks / 32 lanes
    const int r = fi >> 4;
    const int c4 = fi & 15;
    float4 vv = *(const float4*)(ldsf + r * 256 + c4 * 4);
    *(float4*)(out + (size_t)(row0 + r) * DM + c4 * 4) = vv;
  }
}

// ---------------- host launcher ----------------
extern "C" void kernel_launch(void* const* d_in, const int* in_sizes, int n_in,
                              void* d_out, int out_size, void* d_ws, size_t ws_size,
                              hipStream_t stream) {
  (void)in_sizes; (void)n_in; (void)out_size; (void)ws_size;
  const float* qv   = (const float*)d_in[0];
  const float* kvv  = (const float*)d_in[1];
  const float* Wq   = (const float*)d_in[2];
  // d_in[3] = Wk: unused — softmax over a length-1 key axis is identically 1
  const float* Wv   = (const float*)d_in[4];
  const float* inw  = (const float*)d_in[5];
  const float* inb  = (const float*)d_in[6];
  const float* outw = (const float*)d_in[7];
  const float* outb = (const float*)d_in[8];
  const float* Wg   = (const float*)d_in[9];
  const float* bg   = (const float*)d_in[10];
  const float* l1g  = (const float*)d_in[11];
  const float* l1b  = (const float*)d_in[12];
  const float* W1   = (const float*)d_in[13];
  const float* W2   = (const float*)d_in[14];
  const float* l2g  = (const float*)d_in[15];
  const float* l2b  = (const float*)d_in[16];

  unsigned short* ws = (unsigned short*)d_ws;
  size_t o = 0;
  unsigned short* Wq_b = ws + o; o += (size_t)DM * DQ;
  unsigned short* Wg_b = ws + o; o += (size_t)DM * DQ;
  unsigned short* Wv_b = ws + o; o += (size_t)DM * DK;
  unsigned short* Wi_b = ws + o; o += (size_t)DM * DM;
  unsigned short* Wo_b = ws + o; o += (size_t)DM * DM;
  unsigned short* W1_b = ws + o; o += (size_t)DFF * DM;
  unsigned short* W2_b = ws + o; o += (size_t)DM * DFF;

  auto cvt = [&](const float* s, unsigned short* d, int n) {
    cvt_bf16_kernel<<<(n + 255) / 256, 256, 0, stream>>>(s, d, n);
  };
  cvt(Wq, Wq_b, DM * DQ);
  cvt(Wg, Wg_b, DM * DQ);
  cvt(Wv, Wv_b, DM * DK);
  cvt(inw + 2 * DM * DM, Wi_b, DM * DM);   // v-slice of packed in_proj
  cvt(outw, Wo_b, DM * DM);
  cvt(W1, W1_b, DFF * DM);
  cvt(W2, W2_b, DM * DFF);

  hipFuncSetAttribute((const void*)fused_kernel,
                      hipFuncAttributeMaxDynamicSharedMemorySize, SMEM_BYTES);
  fused_kernel<<<B_ROWS / BLOCK_ROWS, 256, SMEM_BYTES, stream>>>(
      qv, kvv, Wq_b, Wg_b, Wv_b, Wi_b, Wo_b, W1_b, W2_b,
      inb, outb, bg, l1g, l1b, l2g, l2b, (float*)d_out);
}
